// MMD_3521873183096
// MI455X (gfx1250) — matrix operand order repaired and checked by reading, not compiled, not run on hardware
//
#include <hip/hip_runtime.h>

typedef _Float16 v4h  __attribute__((ext_vector_type(4)));
typedef _Float16 v8h  __attribute__((ext_vector_type(8)));
typedef _Float16 v16h __attribute__((ext_vector_type(16)));
typedef float    v8f  __attribute__((ext_vector_type(8)));

#define DIMS 64
#define BT   128      // macro-tile edge
#define LDSP 72       // padded LDS row stride in f16 elems (144B, 16B aligned)

// exp(-sq/2) = 2^(-sq * log2(e)/2)
#define HALF_LOG2E 0.72134752044448170f

// ---------------------------------------------------------------------------
__global__ void mmd_zero_kernel(float* accum) {
    if (threadIdx.x < 16) accum[threadIdx.x] = 0.0f;
}

// ---------------------------------------------------------------------------
__global__ void mmd_norms_kernel(const float* __restrict__ X,
                                 const float* __restrict__ Y,
                                 float* __restrict__ nX,
                                 float* __restrict__ nY, int N) {
    int r = blockIdx.x * blockDim.x + threadIdx.x;
    const float* src;
    float* dst;
    int row;
    if (r < N)          { src = X; dst = nX; row = r; }
    else if (r < 2 * N) { src = Y; dst = nY; row = r - N; }
    else return;
    const float4* p = (const float4*)(src + (size_t)row * DIMS);
    float s = 0.f;
#pragma unroll
    for (int q = 0; q < DIMS / 4; ++q) {
        float4 v = p[q];
        s += v.x * v.x + v.y * v.y + v.z * v.z + v.w * v.w;
    }
    dst[row] = s;
}

// ---------------------------------------------------------------------------
// One block = 128x128 output macro-tile of exp(-||x_i - y_j||^2 / 2), fused sum.
// mode 0: rectangular grid (nb*nb blocks).  mode 1: symmetric upper triangle,
// off-diagonal 16x16 tiles weighted 2x (X and Y alias the same matrix).
__global__ __launch_bounds__(256)
void mmd_tile_kernel(const float* __restrict__ X, const float* __restrict__ Y,
                     const float* __restrict__ normX,
                     const float* __restrict__ normY,
                     float* __restrict__ accum, int nb, int mode) {
    __shared__ _Float16 sXh[BT][LDSP];
    __shared__ _Float16 sXl[BT][LDSP];
    __shared__ _Float16 sYh[BT][LDSP];
    __shared__ _Float16 sYl[BT][LDSP];
    __shared__ float    sNx[BT];
    __shared__ float    sNy[BT];
    __shared__ float    wsum[8];

    // ---- decode block index -> (bi, bj) tile-pair ----
    int bi, bj;
    if (mode == 0) {
        bi = blockIdx.x / nb;
        bj = blockIdx.x % nb;
    } else {
        int t = blockIdx.x;
        bi = 0;
        while (t >= nb - bi) { t -= (nb - bi); ++bi; }
        bj = bi + t;
    }

    const int tid = threadIdx.x;

    // ---- stage tiles: fp32 -> (hi f16, lo f16) split in LDS ----
    {
        const int r0 = tid >> 1;          // 0..127
        const int cb = (tid & 1) * 32;    // 0 or 32
        const float4* gx = (const float4*)(X + (size_t)(bi * BT + r0) * DIMS + cb);
        const float4* gy = (const float4*)(Y + (size_t)(bj * BT + r0) * DIMS + cb);
#pragma unroll
        for (int q = 0; q < 8; ++q) {
            float4 vx = gx[q];
            float4 vy = gy[q];
            const int c = cb + q * 4;
            v4h hx, lx, hy, ly;
            hx[0] = (_Float16)vx.x; lx[0] = (_Float16)(vx.x - (float)hx[0]);
            hx[1] = (_Float16)vx.y; lx[1] = (_Float16)(vx.y - (float)hx[1]);
            hx[2] = (_Float16)vx.z; lx[2] = (_Float16)(vx.z - (float)hx[2]);
            hx[3] = (_Float16)vx.w; lx[3] = (_Float16)(vx.w - (float)hx[3]);
            hy[0] = (_Float16)vy.x; ly[0] = (_Float16)(vy.x - (float)hy[0]);
            hy[1] = (_Float16)vy.y; ly[1] = (_Float16)(vy.y - (float)hy[1]);
            hy[2] = (_Float16)vy.z; ly[2] = (_Float16)(vy.z - (float)hy[2]);
            hy[3] = (_Float16)vy.w; ly[3] = (_Float16)(vy.w - (float)hy[3]);
            *(v4h*)&sXh[r0][c] = hx;
            *(v4h*)&sXl[r0][c] = lx;
            *(v4h*)&sYh[r0][c] = hy;
            *(v4h*)&sYl[r0][c] = ly;
        }
        if (tid < BT) sNx[tid] = normX[bi * BT + tid];
        else          sNy[tid - BT] = normY[bj * BT + (tid - BT)];
    }
    __syncthreads();

    const int wave = tid >> 5;
    const int lane = tid & 31;
    const int hl   = lane >> 4;   // half-wave: selects K sub-chunk / M group
    const int mn   = lane & 15;   // A: row M, B: col N

    const int tm = wave;          // this wave's 16-row tile strip (0..7)

    // ---- hoist A fragments (depend only on tm/lane, not tn) ----
    v16h ah[2], al[2];
#pragma unroll
    for (int s = 0; s < 2; ++s) {
        const int arow = tm * 16 + mn;
        const int ka0  = 32 * s + 8 * hl;
        const int ka1  = 32 * s + 16 + 8 * hl;
        v8h ah0 = *(const v8h*)&sXh[arow][ka0];
        v8h ah1 = *(const v8h*)&sXh[arow][ka1];
        v8h al0 = *(const v8h*)&sXl[arow][ka0];
        v8h al1 = *(const v8h*)&sXl[arow][ka1];
        ah[s] = __builtin_shufflevector(ah0, ah1, 0,1,2,3,4,5,6,7,8,9,10,11,12,13,14,15);
        al[s] = __builtin_shufflevector(al0, al1, 0,1,2,3,4,5,6,7,8,9,10,11,12,13,14,15);
    }
    // ---- hoist this wave's 8 row norms (C-layout rows M = v + 8*hl) ----
    float nxv[8];
#pragma unroll
    for (int v = 0; v < 8; ++v) nxv[v] = sNx[tm * 16 + hl * 8 + v];

    float lsum = 0.f;

    for (int tn = 0; tn < 8; ++tn) {
        float wt = 1.0f;
        if (mode == 1) {
            const int gi = bi * 8 + tm;
            const int gj = bj * 8 + tn;
            if (gi > gj) continue;        // wave-uniform skip below diagonal
            wt = (gi == gj) ? 1.0f : 2.0f;
        }
        v8f accH = {};                    // hh chain
        v8f accL = {};                    // hl + lh chain (independent)
#pragma unroll
        for (int s = 0; s < 2; ++s) {     // K steps of 32 over D=64
            const int brow = tn * 16 + mn;
            const int kb   = 32 * s + 8 * hl;
            v8h bh0 = *(const v8h*)&sYh[brow][kb];
            v8h bh1 = *(const v8h*)&sYh[brow][kb + 16];
            v8h bl0 = *(const v8h*)&sYl[brow][kb];
            v8h bl1 = *(const v8h*)&sYl[brow][kb + 16];
            v16h bh = __builtin_shufflevector(bh0, bh1, 0,1,2,3,4,5,6,7,8,9,10,11,12,13,14,15);
            v16h bl = __builtin_shufflevector(bl0, bl1, 0,1,2,3,4,5,6,7,8,9,10,11,12,13,14,15);
            // dot = hh + hl + lh  (drop ll, ~2^-22 relative)
            accH = __builtin_amdgcn_wmma_f32_16x16x32_f16(false, ah[s], false, bh, (short)0, accH, false, false);
            accL = __builtin_amdgcn_wmma_f32_16x16x32_f16(false, ah[s], false, bl, (short)0, accL, false, false);
            accL = __builtin_amdgcn_wmma_f32_16x16x32_f16(false, al[s], false, bh, (short)0, accL, false, false);
        }
        // epilogue: lane col N = mn, VGPR v holds row M = v + 8*hl
        const float ny = sNy[tn * 16 + mn];
        float tsum = 0.f;
#pragma unroll
        for (int v = 0; v < 8; ++v) {
            const float dot = accH[v] + accL[v];
            // -sq/2 * log2(e); clamp to <= 0 (sq >= 0); raw v_exp_f32,
            // hardware underflow->0 is the correct limit for exp(-big)
            float arg = (2.0f * dot - nxv[v] - ny) * HALF_LOG2E;
            arg = fminf(arg, 0.0f);
            tsum += __builtin_amdgcn_exp2f(arg);
        }
        lsum += wt * tsum;
    }

    // ---- block reduction + one atomic per block ----
#pragma unroll
    for (int off = 16; off > 0; off >>= 1)
        lsum += __shfl_xor(lsum, off, 32);
    if (lane == 0) wsum[wave] = lsum;
    __syncthreads();
    if (tid == 0) {
        float s = 0.f;
#pragma unroll
        for (int i = 0; i < 8; ++i) s += wsum[i];
        atomicAdd(accum, s);
    }
}

// ---------------------------------------------------------------------------
__global__ void mmd_finalize_kernel(const float* __restrict__ accum,
                                    float* __restrict__ out, int N) {
    const double Sxy = (double)accum[0];
    const double Sxx = (double)accum[1];   // full symmetric sum incl. diagonal
    const double Syy = (double)accum[2];
    const double n = (double)N;
    // diagonal of XX/YY grams is exp(0)=1 -> trace = N
    const double mmd2 = (Sxx - n) / (n * (n - 1.0))
                      + (Syy - n) / (n * (n - 1.0))
                      - 2.0 * Sxy / (n * n);
    out[0] = (float)(0.5 * mmd2);
}

// ---------------------------------------------------------------------------
extern "C" void kernel_launch(void* const* d_in, const int* in_sizes, int n_in,
                              void* d_out, int out_size, void* d_ws, size_t ws_size,
                              hipStream_t stream) {
    const float* X = (const float*)d_in[0];   // t_data  (N x 64, fp32)
    const float* Y = (const float*)d_in[1];   // f_data  (N x 64, fp32)
    const int N  = in_sizes[0] / DIMS;        // 8192
    const int nb = N / BT;                    // 64 macro-tiles per edge

    float* ws    = (float*)d_ws;
    float* accum = ws;            // [0]=Sxy [1]=Sxx [2]=Syy (16 floats reserved)
    float* nX    = ws + 16;
    float* nY    = nX + N;

    mmd_zero_kernel<<<1, 64, 0, stream>>>(accum);
    mmd_norms_kernel<<<(2 * N + 255) / 256, 256, 0, stream>>>(X, Y, nX, nY, N);

    // XY: full rectangle
    mmd_tile_kernel<<<nb * nb, 256, 0, stream>>>(X, Y, nX, nY, accum + 0, nb, 0);
    // XX, YY: upper-triangle blocks only (off-diagonal tiles weighted 2x)
    const int symBlocks = nb * (nb + 1) / 2;
    mmd_tile_kernel<<<symBlocks, 256, 0, stream>>>(X, X, nX, nX, accum + 1, nb, 1);
    mmd_tile_kernel<<<symBlocks, 256, 0, stream>>>(Y, Y, nY, nY, accum + 2, nb, 1);

    mmd_finalize_kernel<<<1, 1, 0, stream>>>(accum, (float*)d_out, N);
}
